// TorchRNN_68040871903586
// MI455X (gfx1250) — matrix operand order repaired and checked by reading, not compiled, not run on hardware
//
#include <hip/hip_runtime.h>
#include <hip/hip_bf16.h>

// ---- types ----------------------------------------------------------------
typedef _Float16 f16;
typedef __attribute__((ext_vector_type(16))) _Float16 v16h;
typedef __attribute__((ext_vector_type(8)))  _Float16 v8h;
typedef __attribute__((ext_vector_type(8)))  float    v8f;
typedef __attribute__((ext_vector_type(4)))  float    v4f;

#define B_  64
#define S_  2048
#define H_  256
#define LDSH 264   // padded f16 row stride in LDS (264*2/4 = 132 dwords/row -> banks spread)

__device__ __forceinline__ v16h cat8(v8h lo, v8h hi) {
  return __builtin_shufflevector(lo, hi, 0,1,2,3,4,5,6,7,8,9,10,11,12,13,14,15);
}

// A fragment (16x32 f16) for k-chunk kc from row-major f16 W[row][k], ld = row stride.
// ISA layout: lane<16 holds row=lane, K = kc*32 + {0..7, 16..23}; lanes 16-31: +8.
__device__ __forceinline__ v16h load_a_frag(const f16* W, int ld, int mbase, int kc, int lane) {
  int row = mbase + (lane & 15);
  int k0  = kc * 32 + ((lane >> 4) << 3);
  const f16* p = W + row * ld + k0;
  v8h lo = *(const v8h*)p;          // K = k0 .. k0+7
  v8h hi = *(const v8h*)(p + 16);   // K = k0+16 .. k0+23
  return cat8(lo, hi);
}

// B fragment (32x16 f16) for k-chunk kc from row-major f16 M[n][k] (col n = lane&15).
// ISA layout: lane<16 col=lane K=kc*32+{0..7, 16..23}; lanes 16-31 col=lane-16: +8.
__device__ __forceinline__ v16h load_b_frag(const f16* M, int ld, int kc, int lane) {
  int n  = lane & 15;
  int k0 = kc * 32 + ((lane >> 4) << 3);
  const f16* p = M + n * ld + k0;
  v8h lo = *(const v8h*)p;          // K = k0 .. k0+7
  v8h hi = *(const v8h*)(p + 16);   // K = k0+16 .. k0+23
  return cat8(lo, hi);
}

__device__ __forceinline__ v8f wmma_f16(v16h a, v16h b, v8f c) {
  return __builtin_amdgcn_wmma_f32_16x16x32_f16(false, a, false, b, (short)0, c, false, false);
}

// tanh(x) = 1 - 2/(exp(2x)+1), valid for all x (saturates correctly at +/-inf)
__device__ __forceinline__ float fast_tanh(float x) {
  float e = __builtin_amdgcn_exp2f(x * 2.8853900817779268f); // 2*log2(e)*x
  return 1.0f - 2.0f * __builtin_amdgcn_rcpf(e + 1.0f);
}

// ---- phase 0: weights f32 -> f16 ------------------------------------------
__global__ void convert_weights(const float* __restrict__ wih, const float* __restrict__ whh,
                                f16* __restrict__ wih_h, f16* __restrict__ whh_h) {
  int i = blockIdx.x * blockDim.x + threadIdx.x;
  if (i < H_ * H_) {
    wih_h[i] = (f16)wih[i];
    whh_h[i] = (f16)whh[i];
  }
}

// ---- phase 1: xW = x @ W_ih^T, written into d_out[b][s][h] -----------------
// Transposed compute: D[hidden][flatrow] = W_ih_tile(16xK) * xT(Kx16).
__global__ void __launch_bounds__(256) rnn_gemm_xw(const float* __restrict__ x,
                                                   const f16*  __restrict__ wih_h,
                                                   float* __restrict__ xw) {
  __shared__ f16 xsh[16 * LDSH];
  const int tid  = threadIdx.x;
  const int lane = tid & 31;
  const int wv   = tid >> 5;                 // 0..7
  const int rowbase = blockIdx.x * 16;       // flat (b,s) row base

  // stage 16x256 f32 x-tile into LDS as f16 (coalesced)
  const float* xblk = x + (size_t)rowbase * H_;
  for (int i = tid; i < 16 * H_; i += 256) {
    int r = i >> 8, c = i & 255;
    xsh[r * LDSH + c] = (f16)xblk[r * H_ + c];
  }

  // preload A = W_ih fragments, 2 hidden M-tiles per wave (covers 16 tiles total)
  const int mh0 = wv * 16, mh1 = (wv + 8) * 16;
  v16h a0[8], a1[8];
#pragma unroll
  for (int kc = 0; kc < 8; ++kc) {
    a0[kc] = load_a_frag(wih_h, H_, mh0, kc, lane);
    a1[kc] = load_a_frag(wih_h, H_, mh1, kc, lane);
  }
  __syncthreads();

  v8f acc0 = {}; v8f acc1 = {};
  v16h b = load_b_frag(xsh, LDSH, 0, lane);      // B = x^T chunk (shared by both tiles)
#pragma unroll
  for (int kc = 0; kc < 8; ++kc) {
    v16h bn = (kc < 7) ? load_b_frag(xsh, LDSH, kc + 1, lane) : b;  // issue next before consume
    acc0 = wmma_f16(a0[kc], b, acc0);
    acc1 = wmma_f16(a1[kc], b, acc1);
    b = bn;
  }

  // D element (M=hidden mh+hi*8+v, N=flatrow lane&15); hidden contiguous -> b128 stores
  const int hi = lane >> 4, lr = lane & 15;
  float* o0 = xw + (size_t)(rowbase + lr) * H_ + mh0 + hi * 8;
  float* o1 = xw + (size_t)(rowbase + lr) * H_ + mh1 + hi * 8;
  *(v4f*)(o0)     = __builtin_shufflevector(acc0, acc0, 0,1,2,3);
  *(v4f*)(o0 + 4) = __builtin_shufflevector(acc0, acc0, 4,5,6,7);
  *(v4f*)(o1)     = __builtin_shufflevector(acc1, acc1, 0,1,2,3);
  *(v4f*)(o1 + 4) = __builtin_shufflevector(acc1, acc1, 4,5,6,7);
}

// ---- phase 2: sequential scan h = tanh(xw_t + h @ W_hh^T) ------------------
// 4 workgroups x 16 batch rows; 8 waves, wave w owns hidden tiles [16w,..) and
// [16(w+8),..). W_hh lives in VGPRs for all 2048 steps; h is double-buffered
// f16 in LDS so each step needs only ONE workgroup barrier.
__global__ void __launch_bounds__(256) rnn_scan(float* __restrict__ out,     // holds xW, overwritten
                                                const f16* __restrict__ whh_h,
                                                float* __restrict__ hn) {
  __shared__ f16 hsh[2][16 * LDSH];            // h[batch 0..15][hidden 0..255], x2 buffers
  const int tid  = threadIdx.x;
  const int lane = tid & 31;
  const int wv   = tid >> 5;                   // 0..7
  const int lr   = lane & 15;                  // batch col / a-row
  const int hi   = lane >> 4;
  const int bbase = blockIdx.x * 16;
  const int mh0   = wv * 16;                   // hidden tile bases
  const int mh1   = mh0 + 128;

  // preload A = W_hh rows for both tiles (16 fragments = 128 VGPRs)
  v16h aw0[8], aw1[8];
#pragma unroll
  for (int kc = 0; kc < 8; ++kc) {
    aw0[kc] = load_a_frag(whh_h, H_, mh0, kc, lane);
    aw1[kc] = load_a_frag(whh_h, H_, mh1, kc, lane);
  }

  // h0 = 0 (buffer 0 only; buffer 1 is fully written at step 0)
  for (int i = tid; i < 16 * LDSH; i += 256) hsh[0][i] = (f16)0.0f;
  __syncthreads();

  // per-lane pointers: D element (M=hidden mh+hi*8+v, N=batch lr)
  float* op0  = out + ((size_t)(bbase + lr) * S_) * H_ + mh0 + hi * 8;  // tile0 store ptr
  const float* pf0 = op0 + H_;                                          // tile0 prefetch ptr
  float* hnp  = hn + (size_t)(bbase + lr) * H_ + mh0 + hi * 8;

  // prefetch xw for t=0
  v4f c00 = *(const v4f*)(op0);
  v4f c01 = *(const v4f*)(op0 + 4);
  v4f c10 = *(const v4f*)(op0 + 128);
  v4f c11 = *(const v4f*)(op0 + 132);

  f16* hw0 = &hsh[0][lr * LDSH + mh0 + hi * 8];  // h writeback base (buffer 0)
  const int bufstride = 16 * LDSH;

#pragma unroll 1
  for (int t = 0; t < S_; ++t) {
    const f16* hcur = &hsh[t & 1][0];
    f16*       hnxt = hw0 + ((~t) & 1) * bufstride;

    v8f acc0 = __builtin_shufflevector(c00, c01, 0,1,2,3,4,5,6,7);  // C init = xw_t
    v8f acc1 = __builtin_shufflevector(c10, c11, 0,1,2,3,4,5,6,7);

    // prefetch next step's xw (t=S-1 reads land in the h_n tail: in-bounds, discarded)
    c00 = *(const v4f*)(pf0);
    c01 = *(const v4f*)(pf0 + 4);
    c10 = *(const v4f*)(pf0 + 128);
    c11 = *(const v4f*)(pf0 + 132);
    pf0 += H_;

    // K loop with explicit B double-buffer: next chunk's ds loads issue before
    // the WMMA pair consuming the current chunk -> LDS latency hidden.
    v16h b = load_b_frag(hcur, LDSH, 0, lane);
#pragma unroll
    for (int kc = 0; kc < 8; ++kc) {
      v16h bn = (kc < 7) ? load_b_frag(hcur, LDSH, kc + 1, lane) : b;
      acc0 = wmma_f16(aw0[kc], b, acc0);
      acc1 = wmma_f16(aw1[kc], b, acc1);
      b = bn;
    }

    v8f r0, r1;
#pragma unroll
    for (int v = 0; v < 8; ++v) { r0[v] = fast_tanh(acc0[v]); r1[v] = fast_tanh(acc1[v]); }

    // output[b][t][hidden]: 8 consecutive hidden floats per lane -> b128 stores
    *(v4f*)(op0)       = __builtin_shufflevector(r0, r0, 0,1,2,3);
    *(v4f*)(op0 + 4)   = __builtin_shufflevector(r0, r0, 4,5,6,7);
    *(v4f*)(op0 + 128) = __builtin_shufflevector(r1, r1, 0,1,2,3);
    *(v4f*)(op0 + 132) = __builtin_shufflevector(r1, r1, 4,5,6,7);
    op0 += H_;

    // write new h into the other buffer (one ds_store_b128 per tile)
    v8h hh0, hh1;
#pragma unroll
    for (int v = 0; v < 8; ++v) { hh0[v] = (f16)r0[v]; hh1[v] = (f16)r1[v]; }
    *(v8h*)(hnxt)       = hh0;
    *(v8h*)(hnxt + 128) = hh1;

    __syncthreads();   // orders: this step's h-writes before next step's reads,
                       // and last step's reads before this/next step's writes (WAR)

    if (t == S_ - 1) {
      *(v4f*)(hnp)       = __builtin_shufflevector(r0, r0, 0,1,2,3);
      *(v4f*)(hnp + 4)   = __builtin_shufflevector(r0, r0, 4,5,6,7);
      *(v4f*)(hnp + 128) = __builtin_shufflevector(r1, r1, 0,1,2,3);
      *(v4f*)(hnp + 132) = __builtin_shufflevector(r1, r1, 4,5,6,7);
    }
  }
}

// ---- launch ----------------------------------------------------------------
extern "C" void kernel_launch(void* const* d_in, const int* in_sizes, int n_in,
                              void* d_out, int out_size, void* d_ws, size_t ws_size,
                              hipStream_t stream) {
  (void)in_sizes; (void)n_in; (void)out_size; (void)ws_size;
  const float* x   = (const float*)d_in[0];   // (64,2048,256)
  const float* Wih = (const float*)d_in[1];   // (256,256)
  const float* Whh = (const float*)d_in[2];   // (256,256)

  float* out = (float*)d_out;                         // output (B,S,H) then h_n (1,B,H)
  float* hn  = out + (size_t)B_ * S_ * H_;

  f16* wih_h = (f16*)d_ws;                            // 128 KB
  f16* whh_h = wih_h + H_ * H_;                       // 128 KB

  convert_weights<<<(H_ * H_ + 255) / 256, 256, 0, stream>>>(Wih, Whh, wih_h, whh_h);
  rnn_gemm_xw<<<(B_ * S_) / 16, 256, 0, stream>>>(x, wih_h, out);
  rnn_scan<<<B_ / 16, 256, 0, stream>>>(out, whh_h, hn);
}
